// ConvSOM_dense1_10823317586753
// MI455X (gfx1250) — matrix operand order repaired and checked by reading, not compiled, not run on hardware
//
#include <hip/hip_runtime.h>

#define NEG_SLOPE 0.01f
#define INV2S2    0.125f   // 1/(2*sigma^2), sigma = 2
#define CDIM      128
#define HDIM      384
#define P0P1      256
#define NGRAPH    64

typedef __attribute__((ext_vector_type(16))) _Float16      v16h;
typedef __attribute__((ext_vector_type(8)))  float         v8f;
typedef __attribute__((ext_vector_type(4)))  unsigned int  u32x4;

union Frag16 { v16h h; u32x4 q[2]; };

static __device__ __forceinline__ unsigned short f2h_bits(float v) {
  _Float16 h = (_Float16)v;
  return __builtin_bit_cast(unsigned short, h);
}

// async global->LDS copy of 16 bytes (CDNA5 GLOBAL_LOAD_ASYNC_TO_LDS_B128, ASYNCcnt)
static __device__ __forceinline__ void async_copy_b128(unsigned lds_byte_addr,
                                                       const void* gptr) {
  asm volatile("global_load_async_to_lds_b128 %0, %1, off"
               :: "v"(lds_byte_addr), "v"((unsigned long long)(uintptr_t)gptr)
               : "memory");
}
static __device__ __forceinline__ void async_wait0() {
  asm volatile("s_wait_asynccnt 0" ::: "memory");
}

// ---------------- degree / dinv ----------------
__global__ void k_deg_init(float* deg, int n) {
  int i = blockIdx.x * blockDim.x + threadIdx.x;
  if (i < n) deg[i] = 1.0f;                 // self loop contributes 1
}
__global__ void k_deg_scatter(const int* __restrict__ dst, float* deg, int e) {
  int i = blockIdx.x * blockDim.x + threadIdx.x;
  if (i < e) atomicAdd(&deg[dst[i]], 1.0f);
}
__global__ void k_deg_inv(float* deg, int n) {
  int i = blockIdx.x * blockDim.x + threadIdx.x;
  if (i < n) deg[i] = 1.0f / sqrtf(deg[i]); // deg >= 1 always
}

// ---------------- input f32 -> f16 (one pass) ----------------
__global__ void k_x16(const float* __restrict__ x, unsigned short* __restrict__ X16, int n) {
  int tid = blockIdx.x * blockDim.x + threadIdx.x;
  if (tid < n * CDIM) X16[tid] = f2h_bits(x[tid]);
}

// ---------------- B prepack into per-lane WMMA fragment order ----------------
// fragment (n,kc): each lane holds 16 contiguous halves.
// K = kc*32 + 8*(lane>>4) + e (e<8), kc*32 + 16 + 8*(lane>>4) + (e-8) (e>=8) ; N = n*16 + (lane&15)
__global__ void k_prepack_b(const float* __restrict__ B, _Float16* __restrict__ out,
                            int kdim, int ndim, int transpose) {
  int kcN = kdim / 32;
  int total = (ndim / 16) * kcN * 512;
  int tid = blockIdx.x * blockDim.x + threadIdx.x;
  if (tid >= total) return;
  int e    = tid & 15;
  int lane = (tid >> 4) & 31;
  int kc   = (tid >> 9) % kcN;
  int n    = tid / (512 * kcN);
  int K  = kc * 32 + ((e < 8) ? (8 * (lane >> 4) + e) : (16 + 8 * (lane >> 4) + (e - 8)));
  int Nc = n * 16 + (lane & 15);
  float v = transpose ? B[Nc * kdim + K] : B[K * ndim + Nc];
  out[tid] = (_Float16)v;
}

__global__ void k_col_norm(const float* __restrict__ S, float* cn) {
  int p = blockIdx.x * blockDim.x + threadIdx.x;
  if (p < P0P1) {
    float s = 0.f;
    for (int k = 0; k < HDIM; ++k) {
      float t = (float)(_Float16)S[p * HDIM + k];   // f16-rounded, consistent with GEMM
      s += t * t;
    }
    cn[p] = s;
  }
}

// -------- dense GEMM: Out[N,128] = A16[N,:128 of apitch] @ W[128,128] (f16 WMMA) -----
__global__ __launch_bounds__(256) void k_gcn_gemm(const unsigned short* __restrict__ A16,
                                                  int apitch,
                                                  const _Float16* __restrict__ Bpk,
                                                  float* __restrict__ Out, int nrows) {
  __shared__ unsigned short Alds[128 * 136];   // +8-half pad -> conflict-free b128 reads
  const int r0 = blockIdx.x * 128;
  const unsigned lds_base = (unsigned)(uintptr_t)Alds;  // DS offset = low 32 bits
  for (int idx = threadIdx.x; idx < 128 * 16; idx += 256) {
    int r = idx >> 4, k8 = (idx & 15) << 3;
    if (r0 + r < nrows) {
      async_copy_b128(lds_base + (unsigned)(r * 136 + k8) * 2,
                      A16 + (size_t)(r0 + r) * apitch + k8);
    } else {
      u32x4 z = {0u, 0u, 0u, 0u};
      *(u32x4*)&Alds[r * 136 + k8] = z;
    }
  }
  async_wait0();
  __syncthreads();
  const int wave = threadIdx.x >> 5, lane = threadIdx.x & 31;
  const int h = lane >> 4, ln = lane & 15;
  v8f acc[8] = {};
#pragma unroll
  for (int kc = 0; kc < 4; ++kc) {
    Frag16 fa;
    const unsigned short* ap = &Alds[(wave * 16 + ln) * 136 + kc * 32 + h * 8];
    fa.q[0] = *(const u32x4*)ap;
    fa.q[1] = *(const u32x4*)(ap + 16);
#pragma unroll
    for (int n = 0; n < 8; ++n) {
      Frag16 fb;
      const u32x4* bp = (const u32x4*)(Bpk + (((n * 4 + kc) * 32 + lane) << 4));
      fb.q[0] = bp[0];
      fb.q[1] = bp[1];
      acc[n] = __builtin_amdgcn_wmma_f32_16x16x32_f16(false, fa.h, false, fb.h,
                                                      (short)0, acc[n], false, false);
    }
  }
#pragma unroll
  for (int n = 0; n < 8; ++n)
#pragma unroll
    for (int i = 0; i < 8; ++i) {
      int r = r0 + wave * 16 + i + 8 * h;
      if (r < nrows) Out[(size_t)r * CDIM + n * 16 + ln] = acc[n][i];
    }
}

// ---------------- GCN aggregation ----------------
__global__ void k_self_init(const float* __restrict__ hmat, const float* __restrict__ dinv,
                            float* __restrict__ out, int n) {
  int tid = blockIdx.x * blockDim.x + threadIdx.x;
  if (tid >= n * CDIM) return;
  float dv = dinv[tid >> 7];
  out[tid] = dv * dv * hmat[tid];
}
__global__ void k_edge_scatter(const float* __restrict__ hmat, const int* __restrict__ src,
                               const int* __restrict__ dst, const float* __restrict__ dinv,
                               float* out, int e) {
  int tid = blockIdx.x * blockDim.x + threadIdx.x;
  if (tid >= e * 32) return;
  int ed = tid >> 5, c4 = (tid & 31) << 2;
  int s = src[ed], d = dst[ed];
  float nrm = dinv[s] * dinv[d];
  float4 v = *(const float4*)&hmat[(size_t)s * CDIM + c4];
  float* o = &out[(size_t)d * CDIM + c4];
  atomicAdd(o + 0, nrm * v.x);
  atomicAdd(o + 1, nrm * v.y);
  atomicAdd(o + 2, nrm * v.z);
  atomicAdd(o + 3, nrm * v.w);
}
__global__ void k_bias_act(const float* __restrict__ agg, const float* __restrict__ b,
                           unsigned short* __restrict__ H16, int off, int n) {
  int tid = blockIdx.x * blockDim.x + threadIdx.x;
  if (tid >= n * CDIM) return;
  int node = tid >> 7, c = tid & 127;
  float v = agg[tid] + b[c];
  v = (v > 0.f) ? v : NEG_SLOPE * v;
  H16[(size_t)node * HDIM + off + c] = f2h_bits(v);
}

__global__ void k_row_norm(const unsigned short* __restrict__ H16, float* rn, int n) {
  int i = blockIdx.x * blockDim.x + threadIdx.x;
  if (i >= n) return;
  const _Float16* hp = (const _Float16*)(H16 + (size_t)i * HDIM);
  float s = 0.f;
  for (int k = 0; k < HDIM; ++k) { float t = (float)hp[k]; s += t * t; }
  rn[i] = s;
}

// ------- fused SOM: H[64rows,384] @ Sf^T[384,256] + per-row argmin + hs -------
__global__ __launch_bounds__(256) void k_som(const unsigned short* __restrict__ H16,
                                             const _Float16* __restrict__ Spk,
                                             const float* __restrict__ colNorm,
                                             const float* __restrict__ rowNorm,
                                             int* __restrict__ winner,
                                             float* __restrict__ hsArr, int nrows) {
  __shared__ unsigned short Alds[64 * 392];
  __shared__ float candV[64 * 2];
  __shared__ int   candI[64 * 2];
  const int r0 = blockIdx.x * 64;
  const unsigned lds_base = (unsigned)(uintptr_t)Alds;
  for (int idx = threadIdx.x; idx < 64 * 48; idx += 256) {
    int r = idx / 48, k8 = (idx % 48) << 3;
    if (r0 + r < nrows) {
      async_copy_b128(lds_base + (unsigned)(r * 392 + k8) * 2,
                      H16 + (size_t)(r0 + r) * HDIM + k8);
    } else {
      u32x4 z = {0u, 0u, 0u, 0u};
      *(u32x4*)&Alds[r * 392 + k8] = z;
    }
  }
  async_wait0();
  __syncthreads();
  const int wave = threadIdx.x >> 5, lane = threadIdx.x & 31;
  const int h = lane >> 4, ln = lane & 15;
  const int rt = wave & 3, ch = wave >> 2;     // 4 row-tiles x 2 column halves
  v8f acc[8] = {};
#pragma unroll
  for (int kc = 0; kc < 12; ++kc) {
    Frag16 fa;
    const unsigned short* ap = &Alds[(rt * 16 + ln) * 392 + kc * 32 + h * 8];
    fa.q[0] = *(const u32x4*)ap;
    fa.q[1] = *(const u32x4*)(ap + 16);
#pragma unroll
    for (int n = 0; n < 8; ++n) {
      Frag16 fb;
      const u32x4* bp = (const u32x4*)(Spk + ((((ch * 8 + n) * 12 + kc) * 32 + lane) << 4));
      fb.q[0] = bp[0];
      fb.q[1] = bp[1];
      acc[n] = __builtin_amdgcn_wmma_f32_16x16x32_f16(false, fa.h, false, fb.h,
                                                      (short)0, acc[n], false, false);
    }
  }
  float bestV[8]; int bestI[8];
#pragma unroll
  for (int i = 0; i < 8; ++i) { bestV[i] = 3.4e38f; bestI[i] = 0; }
#pragma unroll
  for (int n = 0; n < 8; ++n) {
    int gcol = (ch * 8 + n) * 16 + ln;
    float cn = colNorm[gcol];
#pragma unroll
    for (int i = 0; i < 8; ++i) {
      float v = cn - 2.0f * acc[n][i];     // rowNorm added later (constant per row)
      if (v < bestV[i]) { bestV[i] = v; bestI[i] = gcol; }
    }
  }
#pragma unroll
  for (int off = 1; off < 16; off <<= 1) {
#pragma unroll
    for (int i = 0; i < 8; ++i) {
      float ov = __shfl_xor(bestV[i], off);
      int   oi = __shfl_xor(bestI[i], off);
      if (ov < bestV[i]) { bestV[i] = ov; bestI[i] = oi; }
    }
  }
  if (ln == 0) {
#pragma unroll
    for (int i = 0; i < 8; ++i) {
      int rl = rt * 16 + i + 8 * h;
      candV[rl * 2 + ch] = bestV[i];
      candI[rl * 2 + ch] = bestI[i];
    }
  }
  __syncthreads();
  if (threadIdx.x < 64) {
    int rl = threadIdx.x, gr = r0 + rl;
    if (gr < nrows) {
      float v0 = candV[rl * 2], v1 = candV[rl * 2 + 1];
      int idx = (v0 <= v1) ? candI[rl * 2] : candI[rl * 2 + 1];
      float d2 = fmaxf(rowNorm[gr] + fminf(v0, v1), 0.f);
      winner[gr] = idx;
      hsArr[gr]  = expf(-sqrtf(d2));
    }
  }
}

// ---------------- SOM neighborhood scatter + head ----------------
__global__ void k_g_zero(float* G) {
  int i = blockIdx.x * blockDim.x + threadIdx.x;
  if (i < NGRAPH * P0P1) G[i] = 0.f;
}
__global__ void k_som_scatter(const int* __restrict__ winner, const float* __restrict__ hs,
                              const int* __restrict__ batch, float* G, int n) {
  int tid = blockIdx.x * blockDim.x + threadIdx.x;
  if (tid >= n * P0P1) return;
  int node = tid >> 8, p = tid & 255;
  int w = winner[node];
  int wi = w >> 4, wj = w & 15;
  int i = p >> 4, j = p & 15;
  float di = (float)(i - wi), dj = (float)(j - wj);
  float c = hs[node] * expf(-(di * di + dj * dj) * INV2S2);  // ax*ay collapsed
  atomicAdd(&G[batch[node] * P0P1 + p], c);
}
__global__ void k_final(const float* __restrict__ G, const float* __restrict__ linW,
                        const float* __restrict__ linb, float* out) {
  int g = threadIdx.x;
  if (g < NGRAPH) {
    float s = 0.f;
    for (int p = 0; p < P0P1; ++p) s += G[g * P0P1 + p] * linW[p];
    s += linb[0];
    out[g] = 1.f / (1.f + expf(-s));
  }
}

extern "C" void kernel_launch(void* const* d_in, const int* in_sizes, int n_in,
                              void* d_out, int out_size, void* d_ws, size_t ws_size,
                              hipStream_t stream) {
  (void)n_in; (void)out_size; (void)ws_size;
  const float* x     = (const float*)d_in[0];
  const int*   ei    = (const int*)d_in[1];
  const int*   batch = (const int*)d_in[2];
  const float* W[3]  = {(const float*)d_in[3], (const float*)d_in[5], (const float*)d_in[7]};
  const float* b[3]  = {(const float*)d_in[4], (const float*)d_in[6], (const float*)d_in[8]};
  const float* S     = (const float*)d_in[9];
  const float* linW  = (const float*)d_in[10];
  const float* linb  = (const float*)d_in[11];

  const int N = in_sizes[0] / CDIM;
  const int E = in_sizes[1] / 2;
  const int* src = ei;
  const int* dst = ei + E;

  char* p = (char*)d_ws;
  auto carve = [&](size_t bytes) { char* r = p; p += (bytes + 255) & ~(size_t)255; return r; };
  float*          dinv   = (float*)carve((size_t)N * 4);
  float*          Hlin   = (float*)carve((size_t)N * CDIM * 4);
  float*          Agg    = (float*)carve((size_t)N * CDIM * 4);
  unsigned short* X16    = (unsigned short*)carve((size_t)N * CDIM * 2);
  unsigned short* H16    = (unsigned short*)carve((size_t)N * HDIM * 2);
  float*          rowN   = (float*)carve((size_t)N * 4);
  int*            winner = (int*)carve((size_t)N * 4);
  float*          hsArr  = (float*)carve((size_t)N * 4);
  float*          G      = (float*)carve((size_t)NGRAPH * P0P1 * 4);
  _Float16*       Wpk[3];
  for (int l = 0; l < 3; ++l) Wpk[l] = (_Float16*)carve((size_t)CDIM * CDIM * 2);
  _Float16*       Spk    = (_Float16*)carve((size_t)HDIM * P0P1 * 2);
  float*          colN   = (float*)carve((size_t)P0P1 * 4);

  const int TB = 256;
  auto blk = [](long long t, int bs) { return (unsigned)((t + bs - 1) / bs); };

  k_deg_init   <<<blk(N, TB), TB, 0, stream>>>(dinv, N);
  k_deg_scatter<<<blk(E, TB), TB, 0, stream>>>(dst, dinv, E);
  k_deg_inv    <<<blk(N, TB), TB, 0, stream>>>(dinv, N);
  k_x16        <<<blk((long long)N * CDIM, TB), TB, 0, stream>>>(x, X16, N);
  for (int l = 0; l < 3; ++l)
    k_prepack_b<<<blk(CDIM * CDIM, TB), TB, 0, stream>>>(W[l], Wpk[l], CDIM, CDIM, 0);
  k_prepack_b<<<blk((long long)HDIM * P0P1, TB), TB, 0, stream>>>(S, Spk, HDIM, P0P1, 1);
  k_col_norm <<<1, TB, 0, stream>>>(S, colN);

  for (int l = 0; l < 3; ++l) {
    const unsigned short* a16 = (l == 0) ? X16 : (H16 + (size_t)CDIM * (l - 1));
    const int apitch          = (l == 0) ? CDIM : HDIM;
    k_gcn_gemm    <<<blk(N, 128), TB, 0, stream>>>(a16, apitch, Wpk[l], Hlin, N);
    k_self_init   <<<blk((long long)N * CDIM, TB), TB, 0, stream>>>(Hlin, dinv, Agg, N);
    k_edge_scatter<<<blk((long long)E * 32, TB), TB, 0, stream>>>(Hlin, src, dst, dinv, Agg, E);
    k_bias_act    <<<blk((long long)N * CDIM, TB), TB, 0, stream>>>(Agg, b[l], H16, CDIM * l, N);
  }
  k_row_norm   <<<blk(N, TB), TB, 0, stream>>>(H16, rowN, N);
  k_som        <<<blk(N, 64), TB, 0, stream>>>(H16, Spk, colN, rowN, winner, hsArr, N);
  k_g_zero     <<<blk(NGRAPH * P0P1, TB), TB, 0, stream>>>(G);
  k_som_scatter<<<blk((long long)N * P0P1, TB), TB, 0, stream>>>(winner, hsArr, batch, G, N);
  k_final      <<<1, 64, 0, stream>>>(G, linW, linb, (float*)d_out);
}
